// BaselineFeedforwardNetwork_43396349559493
// MI455X (gfx1250) — matrix operand order repaired and checked by reading, not compiled, not run on hardware
//
#include <hip/hip_runtime.h>

// CDNA5 / gfx1250 (MI455X) — wave32, WMMA 16x16x32 f16 with f32 accumulate.
//
// 60-step sequential scan of a 4->64->64->1 MLP with scalar delta feedback,
// batch N=262144. Compute-bound (~139 GFLOP vs ~252MB traffic -> 10.8us HBM
// floor). Per wave: 16 samples as B-matrix columns, h^T = W^T @ x^T so the
// weight (A) fragments are built once and reused for all 60 steps.
//
// D-tile (16x16 f32) -> next-layer B (32x16 f16) conversion, relu included,
// is done in the packed-f16 domain: cvt_pk -> v_pk_max (relu) -> one
// dword-level select + v_permlanex16 + selects per register pair. The
// layer-3 scalar lands at lane == sample, so the delta recurrence is
// lane-local. bias1 folded into the L1 WMMA via a constant-1.0 K-slot;
// bias2 is the C operand of the first L2 WMMA; layer3 is a VALU dot
// product with 4 independent fmac chains.
//
// B1 is built UNIFORMLY on all lanes: A1 columns k>=5 are exact zeros, so
// the B1 rows held by hi lanes / upper dwords only need to be finite — no
// lane masking, no predicated loads, no VCC dependency in the B1 build.

typedef __attribute__((ext_vector_type(16))) _Float16 v16h;
typedef __attribute__((ext_vector_type(2)))  __fp16   fp16x2;
typedef __attribute__((ext_vector_type(8)))  float    v8f;
typedef __attribute__((ext_vector_type(8)))  int      v8i;

#define T_STEPS 60
#define FEAT    3
#define HIDDEN  64

__device__ __forceinline__ int swap16i(int x) {
    // value from lane ^ 16 (swap the two 16-lane halves of the wave32)
#if __has_builtin(__builtin_amdgcn_permlanex16)
    return __builtin_amdgcn_permlanex16(x, x, 0x76543210, 0xfedcba98,
                                        false, false);
#else
    return __shfl_xor(x, 16, 32);
#endif
}

__device__ __forceinline__ float swap16f(float x) {
    return __builtin_bit_cast(float, swap16i(__builtin_bit_cast(int, x)));
}

// two f32 -> packed f16 dword (round toward zero; fine for activations)
__device__ __forceinline__ int pk2(float x, float y) {
    fp16x2 p = __builtin_amdgcn_cvt_pkrtz(x, y);
    return __builtin_bit_cast(int, p);
}

// packed-f16 relu: single VOP3P op, no canonicalize pair
__device__ __forceinline__ int pkmax0(int x) {
    int r;
    asm("v_pk_max_num_f16 %0, %1, 0" : "=v"(r) : "v"(x));
    return r;
}

// scalar f32 relu: single op, no canonicalize pair
__device__ __forceinline__ float relu(float x) {
    float r;
    asm("v_max_num_f32 %0, %1, 0" : "=v"(r) : "v"(x));
    return r;
}

__global__ __launch_bounds__(256) void mlp_scan_wmma_kernel(
    const float* __restrict__ features,
    const float* __restrict__ W1, const float* __restrict__ b1,
    const float* __restrict__ W2, const float* __restrict__ b2,
    const float* __restrict__ W3, const float* __restrict__ b3,
    float* __restrict__ out, int N)
{
    const int  lane = threadIdx.x & 31;
    const int  wave = threadIdx.x >> 5;
    const bool lo   = lane < 16;
    const int  m    = lane & 15;        // sample-in-wave / matrix column
    const int  hi8  = lo ? 0 : 8;

    const int sampleBase = blockIdx.x * 128 + wave * 16;
    const int s      = sampleBase + m;
    const bool valid = s < N;
    const int  sc    = valid ? s : 0;   // clamp so all lanes load safely

    // ---------------- static A fragments (transposed weights) --------------
    // A 16x32 f16 layout: lane l holds row m=l&15; slot ss holds
    //   k = ss + (ss>=8 ? 8 : 0) + (l>=16 ? 8 : 0)
    v16h A1[4];        // L1: rows = output features 16*ft..16*ft+15 of W1^T,
                       //     k=0..3 -> W1 rows, k=4 -> bias1 (B1 slot4 = 1.0)
    v16h A2[4][2];     // L2: 4 feature tiles x 2 K-chunks of W2^T
#pragma unroll
    for (int ft = 0; ft < 4; ++ft) {
        const int col = 16 * ft + m;   // column of W == output feature
#pragma unroll
        for (int ss = 0; ss < 16; ++ss) {
            const int k = ss + (ss >= 8 ? 8 : 0) + hi8;
            float v = 0.f;
            if (k < 4)       v = W1[k * HIDDEN + col];
            else if (k == 4) v = b1[col];
            A1[ft][ss] = (_Float16)v;
        }
#pragma unroll
        for (int kc = 0; kc < 2; ++kc) {
#pragma unroll
            for (int ss = 0; ss < 16; ++ss) {
                const int k = 32 * kc + ss + 2 * hi8;
                A2[ft][kc][ss] = (_Float16)W2[k * HIDDEN + col];
            }
        }
    }

    // bias2 as WMMA C operand; W3 as per-lane VALU constants.
    // D-tile: vgpr r holds feature 16*ft + r (lanes 0-15) / +r+8 (lanes 16-31)
    v8f bias2v[4];
    v8f w3v[4];
#pragma unroll
    for (int ft = 0; ft < 4; ++ft) {
#pragma unroll
        for (int r = 0; r < 8; ++r) {
            bias2v[ft][r] = b2[16 * ft + r + hi8];
            w3v[ft][r]    = W3[16 * ft + r + hi8];
        }
    }
    const float b3v = b3[0];

    const float* fptr = features + (size_t)sc * (T_STEPS * FEAT);
    float*       optr = out      + (size_t)sc * T_STEPS;

    const v8f zero8 = {0.f, 0.f, 0.f, 0.f, 0.f, 0.f, 0.f, 0.f};

    // B1 dwords: only dwords 0-1 change per step. dword2 = slots {4,5} =
    // {1.0h, 0}: on lo lanes this is the bias-1.0 row (k=4); on hi lanes it
    // maps to k=12 whose A1 column is zero, so it is harmless — uniform.
    v8i b1d;
    b1d[2] = 0x00003C00;
    b1d[3] = 0; b1d[4] = 0; b1d[5] = 0; b1d[6] = 0; b1d[7] = 0;

    float delta = 0.f;

    for (int t = 0; t < T_STEPS; ++t) {
        // ---- B1 = x^T (K=5 used: f0,f1,f2,delta,1.0 ; rest zero/ignored) --
        // Built uniformly on all 32 lanes: rows k>=5 hit zero A1 columns.
        const float f0 = fptr[0], f1 = fptr[1], f2 = fptr[2];
        fptr += FEAT;

        b1d[0] = pk2(f0, f1);
        b1d[1] = pk2(f2, delta);
        const v16h B1 = __builtin_bit_cast(v16h, b1d);

        // ---- layer 1: 4 WMMAs, D = W1aug^T (16x32) x x^T (32x16) ----------
        v8f acc1[4];
#pragma unroll
        for (int ft = 0; ft < 4; ++ft) {
            acc1[ft] = __builtin_amdgcn_wmma_f32_16x16x32_f16(
                false, A1[ft], false, B1, (short)0, zero8, false, false);
        }

        // ---- relu + D-tiles -> B2, all in packed f16 ----------------------
        // Per register pair (a from tile 2kc, b from tile 2kc+1), packed 2 at
        // a time:  slot pair s<8 : lo want a local,        hi want b@(l^16)
        //          slot pair s+8 : lo want a@(l^16),       hi want b local
        // w = lo ? b : a  ==>  swap16(w) = (lo: a@l^16, hi: b@l^16) serves both.
        v16h B2[2];
#pragma unroll
        for (int kc = 0; kc < 2; ++kc) {
            v8i bd;
#pragma unroll
            for (int rp = 0; rp < 4; ++rp) {
                const int pa = pkmax0(pk2(acc1[2*kc  ][2*rp], acc1[2*kc  ][2*rp+1]));
                const int pb = pkmax0(pk2(acc1[2*kc+1][2*rp], acc1[2*kc+1][2*rp+1]));
                const int pw  = lo ? pb : pa;
                const int psw = swap16i(pw);
                bd[rp]     = lo ? pa  : psw;
                bd[4 + rp] = lo ? psw : pb;
            }
            B2[kc] = __builtin_bit_cast(v16h, bd);
        }

        // ---- layer 2: 8 WMMAs (2 K-chunks), bias2 as C operand ------------
        v8f acc2[4];
#pragma unroll
        for (int ft = 0; ft < 4; ++ft) {
            v8f c = __builtin_amdgcn_wmma_f32_16x16x32_f16(
                false, A2[ft][0], false, B2[0], (short)0, bias2v[ft], false, false);
            acc2[ft] = __builtin_amdgcn_wmma_f32_16x16x32_f16(
                false, A2[ft][1], false, B2[1], (short)0, c, false, false);
        }

        // ---- layer 3 (64 -> 1): relu + 4 independent fmac chains ----------
        // lane l holds features {16ft + r + 8*(l>=16)} of sample l&15;
        // lanes l and l^16 cover complementary halves of the 64 features.
        float p[4];
#pragma unroll
        for (int ft = 0; ft < 4; ++ft) {
            p[ft] = relu(acc2[ft][0]) * w3v[ft][0];
#pragma unroll
            for (int r = 1; r < 8; ++r)
                p[ft] = fmaf(relu(acc2[ft][r]), w3v[ft][r], p[ft]);
        }
        const float partial = (p[0] + p[1]) + (p[2] + p[3]);

        delta = (partial + swap16f(partial)) + b3v;  // same on lanes l, l^16

        if (lo && valid) optr[t] = delta;
        // EXEC is restored to all-ones at reconvergence before the next WMMA.
    }
}

extern "C" void kernel_launch(void* const* d_in, const int* in_sizes, int n_in,
                              void* d_out, int out_size, void* d_ws, size_t ws_size,
                              hipStream_t stream) {
    const float* features = (const float*)d_in[0];
    const float* W1 = (const float*)d_in[1];
    const float* b1 = (const float*)d_in[2];
    const float* W2 = (const float*)d_in[3];
    const float* b2 = (const float*)d_in[4];
    const float* W3 = (const float*)d_in[5];
    const float* b3 = (const float*)d_in[6];
    float* out = (float*)d_out;

    const int N = in_sizes[0] / (T_STEPS * FEAT);
    const int blocks = (N + 127) / 128;   // 256 threads = 8 waves x 16 samples

    mlp_scan_wmma_kernel<<<blocks, 256, 0, stream>>>(
        features, W1, b1, W2, b2, W3, b3, out, N);
}
